// SimpleGenerateImpulse_996432413556
// MI455X (gfx1250) — matrix operand order repaired and checked by reading, not compiled, not run on hardware
//
#include <hip/hip_runtime.h>

typedef __attribute__((ext_vector_type(16))) _Float16 v16h;
typedef __attribute__((ext_vector_type(8)))  _Float16 v8h;
typedef __attribute__((ext_vector_type(8)))  float    v8f;

#define ROWS    512     // B * N_EVENTS
#define LATENT  128
#define CH      256
#define NSAMP   32768
#define FSIZE   64
#define ENV_OUT 256

__device__ __forceinline__ v8f wmma_f16(v16h a, v16h b, v8f c) {
  // 8 args: (neg_a, A, neg_b, B, c_mod, C, reuse_a, reuse_b)
  return __builtin_amdgcn_wmma_f32_16x16x32_f16(false, a, false, b, (short)0, c,
                                                false, false);
}

// ---------------------------------------------------------------------------
// Fragment helpers (contiguous k order, same in A and B).
// A 16x32 f16: lanes 0-15 rows M=0..15; lane<16 K=0..15,
//              lane>=16 K=16..31  -> two contiguous 8-half runs.
// B 32x16 f16: lane&15 = column; lane<16 K=0..15, lane>=16 K=16..31.
// C/D f32:     VGPR r -> row r + 8*(lane>=16), col = lane&15.
// ---------------------------------------------------------------------------

// A from LDS f32 tile (row-major, leading dim ld). Requires (kbase%32)==0 and
// rows 32B-aligned so both runs are aligned ds_load_b128 pairs.
__device__ __forceinline__ v16h a_frag_lds_f32(const float* src, int ld,
                                               int kbase, int lane) {
  const int row = lane & 15;
  const int khi = (lane >> 4) << 4;          // 0 or 16
  const float* p = src + row * ld + kbase + khi;
  v8f lo = *(const v8f*)p;                   // K = kbase+khi+0..7
  v8f hi = *(const v8f*)(p + 8);             // K = kbase+khi+8..15
  v16h a;
#pragma unroll
  for (int e = 0; e < 8; ++e) {
    a[e]     = (_Float16)lo[e];
    a[e + 8] = (_Float16)hi[e];
  }
  return a;
}

// A from LDS f16 staging buffer (FIR): same two aligned 8-half runs.
__device__ __forceinline__ v16h a_frag_lds_f16(const _Float16* src, int kbase,
                                               int lane) {
  const int row = lane & 15;
  const int khi = (lane >> 4) << 4;
  const _Float16* p = src + 16 * row + kbase + khi;
  v8h lo = *(const v8h*)p;                   // ds_load_b128
  v8h hi = *(const v8h*)(p + 8);             // ds_load_b128
  v16h a;
#pragma unroll
  for (int e = 0; e < 8; ++e) {
    a[e]     = lo[e];
    a[e + 8] = hi[e];
  }
  return a;
}

// B from pre-packed f16 weights: layout [kBlock][col][32 halfs], so each lane
// reads 16 contiguous halfs (two aligned 16B global loads).
__device__ __forceinline__ v16h b_frag_packed(const _Float16* Wp, int ldn,
                                              int kbase, int col0, int lane) {
  const int col  = col0 + (lane & 15);
  const int koff = (lane >> 4) << 4;         // 0 or 16
  const _Float16* p = Wp + ((size_t)(kbase >> 5) * ldn + col) * 32 + koff;
  v8h lo = *(const v8h*)p;
  v8h hi = *(const v8h*)(p + 8);
  v16h b;
#pragma unroll
  for (int e = 0; e < 8; ++e) {
    b[e]     = lo[e];
    b[e + 8] = hi[e];
  }
  return b;
}

// ---------------------------------------------------------------------------
// Weight pre-pack: (K x N) f32 row-major -> f16 [K/32][N][32] fragment order.
// ---------------------------------------------------------------------------
__global__ __launch_bounds__(256) void pack_w_f16(const float* __restrict__ W,
                                                  _Float16* __restrict__ Wp,
                                                  int K, int N) {
  int i = blockIdx.x * 256 + threadIdx.x;
  if (i >= K * N) return;
  int k = i / N, col = i - k * N;
  Wp[((size_t)(k >> 5) * N + col) * 32 + (k & 31)] = (_Float16)W[i];
}

// ---------------------------------------------------------------------------
// MLP stack: 3 x (Linear + LayerNorm + leaky_relu(0.2)) + final Linear.
// One workgroup (16 waves) owns a 16-row strip; activations live in LDS.
// ---------------------------------------------------------------------------
__global__ __launch_bounds__(512) void mlp_stack_wmma(
    const float* __restrict__ x,
    const _Float16* __restrict__ W0, const float* __restrict__ b0,
    const float* __restrict__ g0, const float* __restrict__ be0,
    const _Float16* __restrict__ W1, const float* __restrict__ b1,
    const float* __restrict__ g1, const float* __restrict__ be1,
    const _Float16* __restrict__ W2, const float* __restrict__ b2,
    const float* __restrict__ g2, const float* __restrict__ be2,
    const _Float16* __restrict__ Wf, const float* __restrict__ bf,
    float* __restrict__ out, int out_dim) {
  __shared__ __align__(32) float hA[16 * CH];
  __shared__ __align__(32) float hB[16 * CH];

  const int tid  = threadIdx.x;
  const int wid  = tid >> 5;
  const int lane = tid & 31;
  const int row0 = blockIdx.x * 16;

  // Stage x strip (16 x 128) into LDS (ld = 256).
  for (int i = tid; i < 16 * LATENT; i += 512) {
    int r = i >> 7, c = i & 127;
    hA[r * CH + c] = x[(row0 + r) * LATENT + c];
  }
  __syncthreads();

  float* cur = hA;
  float* nxt = hB;
  const _Float16* Ws[3] = {W0, W1, W2};
  const float* bs[3]  = {b0, b1, b2};
  const float* gs[3]  = {g0, g1, g2};
  const float* bes[3] = {be0, be1, be2};
  const int    Ks[3]  = {LATENT, CH, CH};

  for (int layer = 0; layer < 3; ++layer) {
    const int K = Ks[layer];
    v8f acc = {};
    for (int kb = 0; kb < K; kb += 32) {
      v16h a = a_frag_lds_f32(cur, CH, kb, lane);
      v16h b = b_frag_packed(Ws[layer], CH, kb, wid * 16, lane);
      acc = wmma_f16(a, b, acc);
    }
    const int   col  = wid * 16 + (lane & 15);
    const int   mhi  = (lane >> 4) << 3;
    const float bias = bs[layer][col];
#pragma unroll
    for (int r = 0; r < 8; ++r)
      nxt[(r + mhi) * CH + col] = acc[r] + bias;
    __syncthreads();

    // LayerNorm + leaky_relu(0.2): wave `wid` owns row `wid` (wave32 reduce).
    float* rowp = nxt + wid * CH;
    float s = 0.f, s2 = 0.f;
#pragma unroll
    for (int c = lane; c < CH; c += 32) {
      float v = rowp[c];
      s += v;
      s2 += v * v;
    }
#pragma unroll
    for (int off = 16; off > 0; off >>= 1) {
      s  += __shfl_xor(s, off, 32);
      s2 += __shfl_xor(s2, off, 32);
    }
    const float m    = s * (1.f / 256.f);
    const float var  = s2 * (1.f / 256.f) - m * m;
    const float rstd = rsqrtf(var + 1e-5f);
#pragma unroll
    for (int c = lane; c < CH; c += 32) {
      float v = (rowp[c] - m) * rstd * gs[layer][c] + bes[layer][c];
      rowp[c] = (v > 0.f) ? v : 0.2f * v;
    }
    __syncthreads();
    float* t = cur; cur = nxt; nxt = t;
  }

  // Final Linear (no norm / activation).
  if (wid * 16 < out_dim) {
    v8f acc = {};
    for (int kb = 0; kb < CH; kb += 32) {
      v16h a = a_frag_lds_f32(cur, CH, kb, lane);
      v16h b = b_frag_packed(Wf, out_dim, kb, wid * 16, lane);
      acc = wmma_f16(a, b, acc);
    }
    const int   col  = wid * 16 + (lane & 15);
    const int   mhi  = (lane >> 4) << 3;
    const float bias = bf[col];
#pragma unroll
    for (int r = 0; r < 8; ++r)
      out[(size_t)(row0 + r + mhi) * out_dim + col] = acc[r] + bias;
  }
}

// ---------------------------------------------------------------------------
// 64-tap FIR via WMMA (fft_convolve with a 64-tap zero-padded filter is a
// linear convolution scaled by 1/sqrt(2N) = 1/256).
// sig[j] = noise[j] * relu(linear_interp(env, j)); computed on the fly to LDS.
// Each wave produces 256 consecutive outputs as one 16x16 tile, K=96:
//   A[m][K'] = sig[i0 + 16m + K' - 63]  -> LDS local index 16m + K' (aligned)
//   B[K'][n] = filt[n + 63 - K'] = frev_pad[K' - n + 16]  (contiguous in e)
// grid = (16 chunks, 512 events), block = 256 threads = 8 waves.
// ---------------------------------------------------------------------------
__global__ __launch_bounds__(256) void fir_wmma(
    const float* __restrict__ noise, const float* __restrict__ env,
    const float* __restrict__ filt, float* __restrict__ out) {
  // Per-wave sig staging: local i <-> j = i0 - 63 + i; valid i in [0,319),
  // rest zero. Row stride 344 halfs = 688B (16B aligned, staggers banks).
  __shared__ __align__(16) _Float16 sig_sh[8][344];
  __shared__ __align__(16) _Float16 frev_pad[128];  // filt[63-i] at i+16

  const int tid  = threadIdx.x;
  const int wid  = tid >> 5;
  const int lane = tid & 31;
  const int ev   = blockIdx.y;
  const int i0   = (blockIdx.x * 8 + wid) * 256;

  const float* envr   = env + (size_t)ev * ENV_OUT;
  const float* noiser = noise + (size_t)ev * NSAMP;

  if (tid < 128) {
    int i = tid - 16;  // frev index
    frev_pad[tid] =
        (i >= 0 && i < FSIZE) ? (_Float16)filt[(size_t)ev * FSIZE + (63 - i)]
                              : (_Float16)0.f;
  }

  // Prefetch next chunk of the noise stream (global_prefetch_b8 path).
  __builtin_prefetch(noiser + ((i0 + 2048) & (NSAMP - 1)) + lane * 8, 0, 1);

  // Stage sig for this wave's tile.
  for (int i = lane; i < 344; i += 32) {
    float v = 0.f;
    int   j = i0 - 63 + i;
    if (i < 319 && j >= 0) {
      // F.interpolate(mode='linear', align_corners=False): L=256 -> 32768
      float pos = ((float)j + 0.5f) * (1.0f / 128.0f) - 0.5f;
      pos = fminf(fmaxf(pos, 0.0f), 255.0f);
      float fi   = floorf(pos);
      int   idx0 = (int)fi;
      int   idx1 = (idx0 + 1 < 255) ? idx0 + 1 : 255;
      float w    = pos - fi;
      float e    = envr[idx0] * (1.0f - w) + envr[idx1] * w;
      e          = fmaxf(e, 0.0f);               // relu AFTER interpolation
      v          = noiser[j] * e;
    }
    sig_sh[wid][i] = (_Float16)v;
  }
  __syncthreads();

  const int col   = lane & 15;
  const int koffB = (lane >> 4) << 4;            // 0 or 16

  v8f acc = {};
#pragma unroll
  for (int kb = 0; kb < 96; kb += 32) {
    v16h a = a_frag_lds_f16(sig_sh[wid], kb, lane);
    v16h b;
    const _Float16* fp = frev_pad + (kb + koffB - col + 16);
#pragma unroll
    for (int e = 0; e < 16; ++e) b[e] = fp[e];   // contiguous run
    acc = wmma_f16(a, b, acc);
  }

  const int mhi  = (lane >> 4) << 3;
  float*    outp = out + (size_t)ev * NSAMP + i0;
#pragma unroll
  for (int r = 0; r < 8; ++r)
    outp[16 * (r + mhi) + col] = acc[r] * (1.0f / 256.0f);
}

// ---------------------------------------------------------------------------

extern "C" void kernel_launch(void* const* d_in, const int* in_sizes, int n_in,
                              void* d_out, int out_size, void* d_ws,
                              size_t ws_size, hipStream_t stream) {
  const float* x     = (const float*)d_in[0];
  const float* noise = (const float*)d_in[1];
  // env_params / filt_params: dict order W0,b0,g0,be0,W1,b1,g1,be1,W2,b2,g2,be2,Wf,bf
  const float* const* ep = (const float* const*)(d_in + 2);
  const float* const* fp = (const float* const*)(d_in + 16);

  float* env_buf  = (float*)d_ws;                 // 512 x 256 f32
  float* filt_buf = env_buf + ROWS * ENV_OUT;     // 512 x 64  f32
  _Float16* wp    = (_Float16*)(filt_buf + ROWS * FSIZE);

  // Packed-weight slots (halfs).
  _Float16* eW0 = wp;                  // 128*256
  _Float16* eW1 = eW0 + LATENT * CH;   // 256*256
  _Float16* eW2 = eW1 + CH * CH;       // 256*256
  _Float16* eWf = eW2 + CH * CH;       // 256*256
  _Float16* fW0 = eWf + CH * ENV_OUT;  // 128*256
  _Float16* fW1 = fW0 + LATENT * CH;   // 256*256
  _Float16* fW2 = fW1 + CH * CH;       // 256*256
  _Float16* fWf = fW2 + CH * CH;       // 256*64

  const int g1 = (LATENT * CH + 255) / 256;   // 128x256 matrices
  const int g2 = (CH * CH + 255) / 256;       // 256x256 matrices
  const int g3 = (CH * FSIZE + 255) / 256;    // 256x64
  pack_w_f16<<<g1, 256, 0, stream>>>(ep[0], eW0, LATENT, CH);
  pack_w_f16<<<g2, 256, 0, stream>>>(ep[4], eW1, CH, CH);
  pack_w_f16<<<g2, 256, 0, stream>>>(ep[8], eW2, CH, CH);
  pack_w_f16<<<g2, 256, 0, stream>>>(ep[12], eWf, CH, ENV_OUT);
  pack_w_f16<<<g1, 256, 0, stream>>>(fp[0], fW0, LATENT, CH);
  pack_w_f16<<<g2, 256, 0, stream>>>(fp[4], fW1, CH, CH);
  pack_w_f16<<<g2, 256, 0, stream>>>(fp[8], fW2, CH, CH);
  pack_w_f16<<<g3, 256, 0, stream>>>(fp[12], fWf, CH, FSIZE);

  mlp_stack_wmma<<<32, 512, 0, stream>>>(
      x, eW0, ep[1], ep[2], ep[3], eW1, ep[5], ep[6], ep[7], eW2, ep[9],
      ep[10], ep[11], eWf, ep[13], env_buf, ENV_OUT);

  mlp_stack_wmma<<<32, 512, 0, stream>>>(
      x, fW0, fp[1], fp[2], fp[3], fW1, fp[5], fp[6], fp[7], fW2, fp[9],
      fp[10], fp[11], fWf, fp[13], filt_buf, FSIZE);

  fir_wmma<<<dim3(16, ROWS), 256, 0, stream>>>(noise, env_buf, filt_buf,
                                               (float*)d_out);
}
